// FGN_81527069212838
// MI455X (gfx1250) — matrix operand order repaired and checked by reading, not compiled, not run on hardware
//
#include <hip/hip_runtime.h>
#include <hip/hip_bf16.h>
#include <stdint.h>

typedef __attribute__((ext_vector_type(16))) __bf16 v16bf;
typedef __attribute__((ext_vector_type(8)))  float  v8f;
typedef __attribute__((ext_vector_type(4)))  unsigned int v4u;
typedef __attribute__((ext_vector_type(8)))  int v8i;
typedef __attribute__((ext_vector_type(4)))  int v4i;

#define PI_D 3.141592653589793238462643383279502884

// ---------------- problem dimensions ----------------
constexpr int B_   = 32;
constexpr int L_   = 32;
constexpr int N_   = 116;
constexpr int E_   = 128;
constexpr int NL   = N_ * L_;        // 3712 (multiple of 64)
constexpr int NF   = NL / 2 + 1;     // 1857 rfft bins
constexpr int NF2  = 2 * NF;         // 3714 interleaved re/im
constexpr int NF2P = 3728;           // NF2 padded to multiple of 16 (GEMM1 N-dim)
constexpr int KG2  = 3776;           // NF2 padded to multiple of 64 (GEMM2 K-dim)
constexpr int NQ   = N_ * 8;         // 928
constexpr int M2   = B_ * E_ * 2;    // 8192
constexpr int D_   = E_ * 2 * 8;     // 2048
constexpr int MR   = B_ * D_;        // 65536
constexpr float LAM = 0.01f;

static inline int cdiv(long a, long b) { return (int)((a + b - 1) / b); }

// ---------------- Tensor Data Mover: 2D bf16 tile -> LDS ----------------
// Builds D# group0/group1 per CDNA5 ISA and issues
// TENSOR_LOAD_TO_LDS. Tile: rows x 16 bf16 elements, row stride = ldb elements.
// Tensor dims are set to the tile region itself (tile start == tensor start),
// so all accesses are in-bounds. Tracked with TENSORcnt.
__device__ inline unsigned int lds_off(const void* p) {
  return (unsigned int)(uintptr_t)p;  // LDS aperture: low 32 bits = LDS offset
}

__device__ inline void tdm_load_tile16(unsigned int lds_addr,
                                       const __bf16* gsrc, int rows, int ldb) {
  const uint64_t ga = (uint64_t)(uintptr_t)gsrc;
  v4u g0;
  g0[0] = 1u;                                  // count=1, user mode, no gather
  g0[1] = lds_addr;                            // lds_addr -> bits [63:32]
  g0[2] = (unsigned int)(ga & 0xFFFFFFFFu);    // global_addr[31:0]
  g0[3] = (unsigned int)((ga >> 32) & 0x1FFFFFFu) | (2u << 30);  // type=2
  const unsigned int tw = 16u;                 // tile_dim0 / tensor_dim0
  const unsigned int th = (unsigned int)rows;  // tile_dim1 / tensor_dim1
  const uint64_t st = (uint64_t)(unsigned int)ldb;  // tensor_dim0_stride (elems)
  v8i g1;
  g1[0] = (int)(1u << 16);                     // data_size=1 (2 bytes)
  g1[1] = (int)((tw & 0xFFFFu) << 16);         // tensor_dim0[15:0] @ bits 63:48
  g1[2] = (int)(((tw >> 16) & 0xFFFFu) | ((th & 0xFFFFu) << 16));   // dim0 hi | dim1 lo
  g1[3] = (int)(((th >> 16) & 0xFFFFu) | ((tw & 0xFFFFu) << 16));   // dim1 hi | tile_dim0
  g1[4] = (int)(th & 0xFFFFu);                 // tile_dim1 (tile_dim2 = 0)
  g1[5] = (int)(st & 0xFFFFFFFFu);             // tensor_dim0_stride[31:0]
  g1[6] = (int)((st >> 32) & 0xFFFFu);         // stride[47:32] (tensor_dim1_stride=0)
  g1[7] = 0;
  v4i z4 = {};
#if defined(__clang_major__) && (__clang_major__ >= 23)
  v8i z8 = {};
  __builtin_amdgcn_tensor_load_to_lds(g0, g1, z4, z4, z8, 0);
#else
  __builtin_amdgcn_tensor_load_to_lds(g0, g1, z4, z4, 0);
#endif
}

// ---------------- WMMA GEMM: TDM-staged B tiles + register-level B reuse -----
// Block: 128 threads (4 waves). Each wave owns MT vertically-stacked 16x16 C
// tiles (block macro-tile = 4*MT*16 rows x 16 cols). B tile (64K x 16N bf16,
// 2KB) staged to LDS by wave 0 via the Tensor Data Mover, double-buffered;
// each wave reads one B fragment per 32-K step (ds ops) and feeds it to MT
// WMMAs with distinct A fragments (global_load_b128 per lane-half), so the
// LDS/VALU packing cost per WMMA drops by MT. K must be a multiple of 64.
// Barrier participation is block-uniform; EXEC is all-ones at every WMMA.
//
// Epilogue modes:
// 0: store f32 (no bias)
// 1: +bias, leaky-relu, store bf16
// 2: +bias, store f32, cols < nlimit only
// 3: G2 scatter: fuse (B,N,E,2,8) -> (B,2048,116-pad-128) transpose, store bf16
template <int MODE, int MT>
__global__ void wmma_gemm(const __bf16* __restrict__ A, int lda,
                          const __bf16* __restrict__ Bm, int ldb,
                          const float* __restrict__ bias,
                          float* __restrict__ outF, __bf16* __restrict__ outB,
                          int ldc, int M, int N, int K, int nlimit) {
  __shared__ __bf16 Bs[2][64 * 16];
  const int wave = threadIdx.x >> 5;
  const int lane = threadIdx.x & 31;
  const int half = lane >> 4;   // 0: lanes 0-15, 1: lanes 16-31
  const int l16  = lane & 15;
  const int n0 = blockIdx.y * 16;
  const int mbase = (blockIdx.x * 4 + wave) * (MT * 16);
  const bool active = (mbase < M);
  const int mc = active ? mbase : 0;     // clamp so inactive waves stay in-bounds
  const int aoff = half * 16;            // A K sub-offset (contiguous, matches B)
  const int boff = half * 16;            // B K sub-offset (contiguous, matches A)

  if (wave == 0) {
    tdm_load_tile16(lds_off(&Bs[0][0]), Bm + n0, 64, ldb);
    __builtin_amdgcn_s_wait_tensorcnt(0);
  }
  __syncthreads();

  v8f acc[MT];
#pragma unroll
  for (int t = 0; t < MT; ++t) acc[t] = (v8f){};

  const int nsteps = K >> 6;
  for (int s = 0; s < nsteps; ++s) {
    const int buf = s & 1;
    if (wave == 0 && (s + 1) < nsteps) {  // prefetch next 64-K B tile via TDM
      tdm_load_tile16(lds_off(&Bs[buf ^ 1][0]),
                      Bm + (long)(s + 1) * 64 * ldb + n0, 64, ldb);
    }
#pragma unroll
    for (int hh = 0; hh < 2; ++hh) {
      const int k0 = (s << 6) + (hh << 5);
      // One B fragment per 32-K step, shared by all MT WMMAs of this wave.
      const __bf16* bsrc = &Bs[buf][((hh << 5) + boff) * 16 + l16];
      v16bf b;
#pragma unroll
      for (int i = 0; i < 16; ++i) b[i] = bsrc[i * 16];   // ds loads
#pragma unroll
      for (int t = 0; t < MT; ++t) {
        const __bf16* ap = A + (long)(mc + t * 16 + l16) * lda + k0 + aoff;
        __builtin_prefetch(ap + 128, 0, 1);
        v16bf a;
#pragma unroll
        for (int i = 0; i < 8; ++i) a[i] = ap[i];           // K = aoff+0..7
#pragma unroll
        for (int i = 0; i < 8; ++i) a[8 + i] = ap[8 + i];   // K = aoff+8..15
        acc[t] = __builtin_amdgcn_wmma_f32_16x16x32_bf16(false, a, false, b,
                                                         (short)0, acc[t],
                                                         false, false);
      }
    }
    if (wave == 0) __builtin_amdgcn_s_wait_tensorcnt(0);
    __syncthreads();
  }

  if (!active) return;
  const int col = n0 + l16;
#pragma unroll
  for (int t = 0; t < MT; ++t) {
    const int rbase = mbase + t * 16 + (half << 3);
#pragma unroll
    for (int v = 0; v < 8; ++v) {
      const int row = rbase + v;
      float x = acc[t][v];
      if (MODE == 0) {
        outF[(long)row * ldc + col] = x;
      } else if (MODE == 1) {
        x += bias[col];
        x = (x >= 0.f) ? x : 0.01f * x;
        outB[(long)row * ldc + col] = (__bf16)x;
      } else if (MODE == 2) {
        if (col < nlimit) outF[(long)row * ldc + col] = x + bias[col];
      } else {  // MODE 3
        const int b = row >> 8, e = (row >> 1) & 127, c = row & 1;
        const int n = col >> 3, j = col & 7;
        const long drow = (long)b * D_ + e * 16 + c * 8 + j;
        outB[drow * 128 + n] = (__bf16)x;
      }
    }
  }
}

// ---------------- fill / elementwise kernels ----------------

// h[b, n*L + l] = x[b, l, n]  (bf16)
__global__ void k_build_hb(const float* __restrict__ x, __bf16* __restrict__ hb) {
  long i = (long)blockIdx.x * blockDim.x + threadIdx.x;
  if (i >= (long)B_ * NL) return;
  int t = (int)(i % NL), b = (int)(i / NL);
  int n = t / L_, l = t % L_;
  hb[i] = (__bf16)x[((long)b * L_ + l) * N_ + n];
}

// Forward ortho-DFT basis: W[t, 2f] = cos(-2pi f t/NL)/sqrt(NL), W[t,2f+1] = sin(-..)
__global__ void k_fill_Wf(__bf16* __restrict__ Wf) {
  long i = (long)blockIdx.x * blockDim.x + threadIdx.x;
  if (i >= (long)NL * NF2P) return;
  int col = (int)(i % NF2P), t = (int)(i / NF2P);
  float v = 0.f;
  if (col < NF2) {
    int f = col >> 1;
    double a = -2.0 * PI_D * (double)(((long)f * t) % NL) / (double)NL;
    v = (col & 1) ? (float)sin(a) : (float)cos(a);
    v *= rsqrtf((float)NL);
  }
  Wf[i] = (__bf16)v;
}

// Combined iRFFT + embeddings_10 basis:
// T[2f+ri, n*8+j] = w_f/sqrt(NL) * sum_l basis(f, n*L+l, ri) * e10[l, j]
__global__ void k_fill_T(const float* __restrict__ e10, __bf16* __restrict__ Tm) {
  long i = (long)blockIdx.x * blockDim.x + threadIdx.x;
  if (i >= (long)KG2 * NQ) return;
  int q = (int)(i % NQ), k = (int)(i / NQ);
  float v = 0.f;
  if (k < NF2) {
    int f = k >> 1, ri = k & 1;
    int n = q >> 3, j = q & 7;
    float w = (f == 0 || f == NF - 1) ? (ri ? 0.f : 1.f) : 2.f;
    if (w != 0.f) {
      float acc = 0.f;
      for (int l = 0; l < L_; ++l) {
        long t = (long)n * L_ + l;
        double a = 2.0 * PI_D * (double)(((long)f * t) % NL) / (double)NL;
        float basis = ri ? (float)(-sin(a)) : (float)cos(a);
        acc += basis * e10[l * 8 + j];
      }
      v = acc * w * rsqrtf((float)NL);
    }
  }
  Tm[i] = (__bf16)v;
}

__device__ inline float softshrinkf(float v) {
  return (fabsf(v) > LAM) ? (v - copysignf(LAM, v)) : 0.f;
}

// 6 complex diagonal stages; writes G rows (b,e,c) with re/im interleaved over f.
__global__ void k_stages(const float* __restrict__ Hc, const float* __restrict__ emb,
                         const float* w1, const float* s1,
                         const float* w2, const float* s2,
                         const float* w3, const float* s3,
                         const float* w4, const float* s4,
                         const float* w5, const float* s5,
                         const float* w6, const float* s6,
                         __bf16* __restrict__ Ag) {
  long i = (long)blockIdx.x * blockDim.x + threadIdx.x;
  if (i >= (long)B_ * NF * E_) return;
  int e = (int)(i % E_);
  long r = i / E_;
  int f = (int)(r % NF);
  int b = (int)(r / NF);
  const float sc = emb[e];
  const float hr = sc * Hc[(long)b * NF2P + 2 * f];
  const float hi = sc * Hc[(long)b * NF2P + 2 * f + 1];
  const float* ws[6] = {w1, w2, w3, w4, w5, w6};
  const float* bs[6] = {s1, s2, s3, s4, s5, s6};
  const long de = (long)e * (E_ + 1) * 2;  // ((i*E+e)*E+e)*2 base for i=0
  float xr = hr, xi = hi;
  float opr0 = 0.f, opr1 = 0.f, opi0 = 0.f, opi1 = 0.f;
#pragma unroll
  for (int s = 0; s < 6; ++s) {
    const float* w = ws[s];
    const float* bp = bs[s];
    float nr = 0.f, ni = 0.f;
#pragma unroll
    for (int c = 0; c < 2; ++c) {
      const float d0 = w[de + c];
      const float d1 = w[(long)E_ * E_ * 2 + de + c];
      const float b0 = bp[(long)e * 2 + c];
      const float b1 = bp[(long)E_ * 2 + (long)e * 2 + c];
      float re = xr * d0 - xi * d1 + b0;
      float im = xi * d0 + xr * d1 + b1;
      re = fmaxf(re, 0.f);
      im = fmaxf(im, 0.f);
      re = softshrinkf(re);
      im = softshrinkf(im);
      if (c == 0) { opr0 += re; opi0 += im; } else { opr1 += re; opi1 += im; }
      nr += re;
      ni += im;
    }
    xr = nr;
    xi = ni;
  }
  const long row0 = ((long)(b * E_ + e)) * 2;
  Ag[row0 * KG2 + 2 * f]           = (__bf16)(opr0 + hr);
  Ag[row0 * KG2 + 2 * f + 1]       = (__bf16)(opi0 + hi);
  Ag[(row0 + 1) * KG2 + 2 * f]     = (__bf16)(opr1 + hr);
  Ag[(row0 + 1) * KG2 + 2 * f + 1] = (__bf16)(opi1 + hi);
}

// zero columns [NF2, KG2) of Ag (K padding)
__global__ void k_pad_Ag(__bf16* __restrict__ Ag) {
  long i = (long)blockIdx.x * blockDim.x + threadIdx.x;
  const int padw = KG2 - NF2;  // 62
  if (i >= (long)M2 * padw) return;
  long row = i / padw;
  int c = (int)(i % padw);
  Ag[row * KG2 + NF2 + c] = (__bf16)0.f;
}

__global__ void k_zero_bf16(__bf16* __restrict__ p, long n) {
  long i = (long)blockIdx.x * blockDim.x + threadIdx.x;
  if (i < n) p[i] = (__bf16)0.f;
}

// dst(Kp x Np, row-major) = src(Nsrc x Ksrc, row-major) transposed + zero-padded, bf16
__global__ void k_wtrans(const float* __restrict__ src, __bf16* __restrict__ dst,
                         int Ksrc, int Nsrc, int Kp, int Np) {
  long i = (long)blockIdx.x * blockDim.x + threadIdx.x;
  if (i >= (long)Kp * Np) return;
  int n = (int)(i % Np), k = (int)(i / Np);
  float v = (k < Ksrc && n < Nsrc) ? src[(long)n * Ksrc + k] : 0.f;
  dst[i] = (__bf16)v;
}

// rW3 layer: per-row dot of length 64, +rb3, no activation, bf16 out
__global__ void k_dot64(const __bf16* __restrict__ A4, const float* __restrict__ rW3,
                        const float* __restrict__ rb3, __bf16* __restrict__ z) {
  long r = (long)blockIdx.x * blockDim.x + threadIdx.x;
  if (r >= (long)MR) return;
  float acc = rb3[0];
#pragma unroll
  for (int k = 0; k < 64; ++k) acc += (float)A4[r * 64 + k] * rW3[k];
  z[r] = (__bf16)acc;
}

// ---------------- launch ----------------
extern "C" void kernel_launch(void* const* d_in, const int* in_sizes, int n_in,
                              void* d_out, int out_size, void* d_ws, size_t ws_size,
                              hipStream_t stream) {
  const float* x    = (const float*)d_in[0];
  const float* emb  = (const float*)d_in[1];
  const float* e10  = (const float*)d_in[2];
  const float* w[6];
  const float* bb[6];
  for (int i = 0; i < 6; ++i) { w[i] = (const float*)d_in[3 + 2 * i]; bb[i] = (const float*)d_in[4 + 2 * i]; }
  const float* rW1 = (const float*)d_in[15];
  const float* rb1 = (const float*)d_in[16];
  const float* rW2 = (const float*)d_in[17];
  const float* rb2 = (const float*)d_in[18];
  const float* rW3 = (const float*)d_in[19];
  const float* rb3 = (const float*)d_in[20];
  const float* fW1 = (const float*)d_in[21];
  const float* fb1 = (const float*)d_in[22];
  const float* fW2 = (const float*)d_in[23];
  const float* fb2 = (const float*)d_in[24];
  const float* fW3 = (const float*)d_in[25];
  const float* fb3 = (const float*)d_in[26];
  float* out = (float*)d_out;

  // workspace carve-up (256B aligned)
  char* ws = (char*)d_ws;
  size_t off = 0;
  auto alloc = [&](size_t bytes) {
    void* p = ws + off;
    off = (off + bytes + 255) & ~(size_t)255;
    return p;
  };
  __bf16* hb   = (__bf16*)alloc((size_t)B_ * NL * 2);
  __bf16* Wf   = (__bf16*)alloc((size_t)NL * NF2P * 2);
  float*  Hc   = (float*) alloc((size_t)B_ * NF2P * 4);
  __bf16* Ag   = (__bf16*)alloc((size_t)M2 * KG2 * 2);
  __bf16* Tm   = (__bf16*)alloc((size_t)KG2 * NQ * 2);
  __bf16* A2   = (__bf16*)alloc((size_t)MR * 128 * 2);
  __bf16* A3   = (__bf16*)alloc((size_t)MR * 256 * 2);
  __bf16* A4   = (__bf16*)alloc((size_t)MR * 64 * 2);
  __bf16* zb   = (__bf16*)alloc((size_t)MR * 2);
  __bf16* rW1t = (__bf16*)alloc((size_t)128 * 256 * 2);
  __bf16* rW2t = (__bf16*)alloc((size_t)256 * 64 * 2);
  __bf16* fW1t = (__bf16*)alloc((size_t)2048 * 128 * 2);
  __bf16* fW2t = (__bf16*)alloc((size_t)128 * 32 * 2);
  __bf16* fW3t = (__bf16*)alloc((size_t)64 * 16 * 2);
  __bf16* y1   = (__bf16*)alloc((size_t)32 * 128 * 2);
  __bf16* y2   = (__bf16*)alloc((size_t)32 * 64 * 2);
  (void)ws_size; (void)n_in; (void)in_sizes; (void)out_size;

  const int TB = 256;
  // --- basis / weight prep ---
  k_build_hb<<<cdiv((long)B_ * NL, TB), TB, 0, stream>>>(x, hb);
  k_fill_Wf<<<cdiv((long)NL * NF2P, TB), TB, 0, stream>>>(Wf);
  k_fill_T<<<cdiv((long)KG2 * NQ, TB), TB, 0, stream>>>(e10, Tm);
  k_wtrans<<<cdiv((long)128 * 256, TB), TB, 0, stream>>>(rW1, rW1t, 116, 256, 128, 256);
  k_wtrans<<<cdiv((long)256 * 64, TB), TB, 0, stream>>>(rW2, rW2t, 256, 64, 256, 64);
  k_wtrans<<<cdiv((long)2048 * 128, TB), TB, 0, stream>>>(fW1, fW1t, 2048, 128, 2048, 128);
  k_wtrans<<<cdiv((long)128 * 32, TB), TB, 0, stream>>>(fW2, fW2t, 128, 32, 128, 32);
  k_wtrans<<<cdiv((long)64 * 16, TB), TB, 0, stream>>>(fW3, fW3t, 32, 2, 64, 16);
  k_pad_Ag<<<cdiv((long)M2 * (KG2 - NF2), TB), TB, 0, stream>>>(Ag);
  k_zero_bf16<<<cdiv((long)MR * 128, TB), TB, 0, stream>>>(A2, (long)MR * 128);
  k_zero_bf16<<<cdiv((long)32 * 64, TB), TB, 0, stream>>>(y2, (long)32 * 64);

  dim3 blk(128);
  // G1: forward DFT  H(32 x 3728) = hb(32 x 3712) * Wf         (MT=1, M=32)
  wmma_gemm<0, 1><<<dim3(1, NF2P / 16), blk, 0, stream>>>(
      hb, NL, Wf, NF2P, nullptr, Hc, nullptr, NF2P, B_, NF2P, NL, 0);
  // stages (elementwise)
  k_stages<<<cdiv((long)B_ * NF * E_, TB), TB, 0, stream>>>(
      Hc, emb, w[0], bb[0], w[1], bb[1], w[2], bb[2], w[3], bb[3], w[4], bb[4], w[5], bb[5], Ag);
  // G2: fused iRFFT + emb10 contraction + layout scatter -> A2  (MT=4)
  wmma_gemm<3, 4><<<dim3(M2 / 256, NQ / 16), blk, 0, stream>>>(
      Ag, KG2, Tm, NQ, nullptr, nullptr, A2, 128, M2, NQ, KG2, 0);
  // G3: rW1 layer -> A3(65536 x 256)                            (MT=4)
  wmma_gemm<1, 4><<<dim3(MR / 256, 256 / 16), blk, 0, stream>>>(
      A2, 128, rW1t, 256, rb1, nullptr, A3, 256, MR, 256, 128, 0);
  // G4: rW2 layer -> A4(65536 x 64)                             (MT=4)
  wmma_gemm<1, 4><<<dim3(MR / 256, 64 / 16), blk, 0, stream>>>(
      A3, 256, rW2t, 64, rb2, nullptr, A4, 64, MR, 64, 256, 0);
  // rW3 layer (N=1 dot)
  k_dot64<<<cdiv((long)MR, TB), TB, 0, stream>>>(A4, rW3, rb3, zb);
  // G6: fW1 layer  y1(32 x 128)                                 (MT=1)
  wmma_gemm<1, 1><<<dim3(1, 128 / 16), blk, 0, stream>>>(
      zb, 2048, fW1t, 128, fb1, nullptr, y1, 128, B_, 128, 2048, 0);
  // G7: fW2 layer  y2(32 x 32, ld 64; cols 32..63 pre-zeroed)   (MT=1)
  wmma_gemm<1, 1><<<dim3(1, 32 / 16), blk, 0, stream>>>(
      y1, 128, fW2t, 32, fb2, nullptr, y2, 64, B_, 32, 128, 0);
  // G8: fW3 layer -> out(32 x 2) f32   (K padded to 64)         (MT=1)
  wmma_gemm<2, 1><<<dim3(1, 1), blk, 0, stream>>>(
      y2, 64, fW3t, 16, fb3, out, nullptr, 2, B_, 16, 64, 2);
}
